// TopoformerLayer_74225624809855
// MI455X (gfx1250) — matrix operand order repaired and checked by reading, not compiled, not run on hardware
//
#include <hip/hip_runtime.h>
#include <math.h>

#define DEVINL __device__ __forceinline__

typedef __attribute__((ext_vector_type(16))) __bf16 v16bf;
typedef __attribute__((ext_vector_type(8)))  float  v8f;

constexpr int BB  = 16;    // batch
constexpr int SS  = 512;   // sequence
constexpr int DD  = 768;   // model dim
constexpr int HH  = 12;    // heads
constexpr int HDD = 64;    // head dim
constexpr int KNN = 16;    // top-k
constexpr int NSMP= 128;   // LSH samples
constexpr int MM  = BB*SS;       // 8192 rows
constexpr int D3  = 3*DD;        // 2304
constexpr int DFF = 2*DD;        // 1536

// ---------------- bf16 helpers ----------------
DEVINL unsigned short f2bf(float f){
  unsigned int u = __float_as_uint(f);
  u += 0x7FFFu + ((u >> 16) & 1u);          // round-to-nearest-even
  return (unsigned short)(u >> 16);
}

union Frag16 { v16bf v; uint4 q[2]; };

// B fragment: lane holds column n = lane&15, 16 contiguous K starting at 16*half.
// Bt is [N][K] row-major bf16; p points at &Bt[n][k0 + 16*half].
DEVINL v16bf load_b_frag(const unsigned short* p){
  Frag16 u;
  u.q[0] = *(const uint4*)(p);
  u.q[1] = *(const uint4*)(p + 8);
  return u.v;
}

// A fragment: lane holds row m = lane&15; elems 0..15 = K[k+16*half .. +15].
DEVINL v16bf load_a_frag(const unsigned short* row, int k, int half){
  Frag16 u;
  u.q[0] = *(const uint4*)(row + k + 16*half);
  u.q[1] = *(const uint4*)(row + k + 16*half + 8);
  return u.v;
}

DEVINL v8f wmma_bf16(v16bf a, v16bf b, v8f c){
  return __builtin_amdgcn_wmma_f32_16x16x32_bf16(false, a, false, b, (short)0, c, false, false);
}

// ---------------- prep kernels ----------------
__global__ __launch_bounds__(256) void k_cvt_x(const float* __restrict__ x,
                                               unsigned short* __restrict__ xb,
                                               float* __restrict__ xsq){
  int row = blockIdx.x, tid = threadIdx.x;
  const float* xr = x + (size_t)row*DD;
  unsigned short* ob = xb + (size_t)row*DD;
  float s = 0.f;
  for (int d = tid; d < DD; d += 256){ float v = xr[d]; ob[d] = f2bf(v); s += v*v; }
  __shared__ float red[256];
  red[tid] = s; __syncthreads();
  for (int st = 128; st > 0; st >>= 1){ if (tid < st) red[tid] += red[tid+st]; __syncthreads(); }
  if (tid == 0) xsq[row] = red[0];
}

__global__ __launch_bounds__(256) void k_gather_s(const unsigned short* __restrict__ xb,
                                                  const float* __restrict__ xsq,
                                                  const int* __restrict__ sidx,
                                                  unsigned short* __restrict__ sb,
                                                  float* __restrict__ ssq){
  int row = blockIdx.x;              // b*NSMP + m
  int b = row / NSMP, m = row % NSMP;
  int src = b*SS + sidx[m];
  const unsigned short* s = xb + (size_t)src*DD;
  unsigned short* d = sb + (size_t)row*DD;
  for (int i = threadIdx.x; i < DD; i += 256) d[i] = s[i];
  if (threadIdx.x == 0) ssq[row] = xsq[src];
}

// w [Kd][Nd] f32 row-major  ->  wt [Nd][Kd] bf16 row-major
__global__ __launch_bounds__(256) void k_transpose(const float* __restrict__ w,
                                                   unsigned short* __restrict__ wt,
                                                   int Kd, int Nd){
  size_t i = (size_t)blockIdx.x*256 + threadIdx.x;
  if (i >= (size_t)Kd*Nd) return;
  int k = (int)(i / Nd), n = (int)(i % Nd);
  wt[(size_t)n*Kd + k] = f2bf(w[i]);
}

__global__ void k_init_acc(float* s, float* q, unsigned int* mn, unsigned int* mx){
  int t = threadIdx.x;
  if (t < BB){ s[t] = 0.f; q[t] = 0.f; mn[t] = 0x7F800000u; mx[t] = 0u; }
}

// ---------------- generic single-wave WMMA GEMM (16x64 tile) ----------------
// C[m0:m0+16, n0:n0+64] = A[m,:K] * Bt[n,:K]^T  (+ mode-specific epilogue)
struct GemmP {
  const unsigned short* A;   // [M][lda] bf16
  const unsigned short* Bt;  // [N][ldb] bf16 (weight^T)
  int Kdim, lda, ldb;
  const float* bias;         // [N]
  const float* add;          // residual [M][768]
  const float* topo;         // [B][768] (pre-scaled by gate)
  const float* xsq;          // [M]
  const float* ssq;          // [B*NSMP]
  float* outF;
  unsigned short* outB;
  unsigned short *q, *k, *v; // QKV outputs
  float* dist;               // [M][NSMP]
};

// MODE: 0=QKV  1=OUT(+x+topo)  2=FFN1(gelu->bf16)  3=FFN2(+x1)  4=DIST
template<int MODE>
__global__ __launch_bounds__(32) void k_gemm(GemmP p){
  const int lane = threadIdx.x;
  const int c    = lane & 15;
  const int half = lane >> 4;
  const int m0   = blockIdx.x * 16;
  const int n0   = blockIdx.y * 64;

  v8f acc[4] = {};
  const unsigned short* Arow = p.A + (size_t)(m0 + c) * p.lda;
  const unsigned short* Bb   = p.Bt;
  if (MODE == 4) Bb += (size_t)(m0 / SS) * NSMP * p.ldb;   // per-batch sample block

  const size_t bstride = (size_t)16 * p.ldb;
  for (int k = 0; k < p.Kdim; k += 32){
    v16bf a = load_a_frag(Arow, k, half);
    const unsigned short* B0 = Bb + (size_t)(n0 + c) * p.ldb + k + half*16;
    #pragma unroll
    for (int j = 0; j < 4; ++j)
      acc[j] = wmma_bf16(a, load_b_frag(B0 + (size_t)j * bstride), acc[j]);
  }

  #pragma unroll
  for (int j = 0; j < 4; ++j){
    int gn = n0 + j*16 + c;
    #pragma unroll
    for (int r = 0; r < 8; ++r){
      int gm = m0 + r + 8*half;
      float val = acc[j][r];
      if (MODE == 0){
        float vb = val + p.bias[gn];
        int which = gn / DD, d = gn % DD;
        int hh = d >> 6, hd = d & 63;
        int b = gm >> 9, s = gm & 511;
        size_t bh = (size_t)(b*HH + hh);
        unsigned short bf = f2bf(vb);
        if      (which == 0) p.q[(bh*SS + s)*HDD + hd] = bf;
        else if (which == 1) p.k[(bh*SS + s)*HDD + hd] = bf;
        else                 p.v[(bh*HDD + hd)*SS + s] = bf;   // V stored transposed
      } else if (MODE == 1){
        float vb = val + p.bias[gn] + p.add[(size_t)gm*DD + gn]
                 + p.topo[(size_t)(gm >> 9)*DD + gn];
        p.outF[(size_t)gm*DD + gn] = vb;
      } else if (MODE == 2){
        float vb = val + p.bias[gn];
        float g  = 0.5f * vb * (1.0f + erff(vb * 0.70710678118654752f));
        p.outB[(size_t)gm*DFF + gn] = f2bf(g);
      } else if (MODE == 3){
        float vb = val + p.bias[gn] + p.add[(size_t)gm*DD + gn];
        p.outF[(size_t)gm*DD + gn] = vb;
      } else { // DIST
        float d2 = p.xsq[gm] + p.ssq[(size_t)(gm >> 9)*NSMP + gn] - 2.0f*val;
        p.dist[(size_t)gm*NSMP + gn] = sqrtf(fmaxf(d2, 0.0f));
      }
    }
  }
}

// ---------------- top-16-of-128 + per-batch stats accumulation ----------------
__global__ __launch_bounds__(128) void k_topk(const float* __restrict__ dist,
                                              float* accS, float* accQ,
                                              unsigned int* accMn, unsigned int* accMx){
  int row = blockIdx.x, tid = threadIdx.x;
  int b = row / SS;
  __shared__ float sv[128];
  __shared__ int   si[128];
  float v = dist[(size_t)row*NSMP + tid];
  float lsum = 0.f, lsq = 0.f, lmax = 0.f, lmin = 0.f;
  for (int t = 0; t < KNN; ++t){
    sv[tid] = v; si[tid] = tid; __syncthreads();
    for (int st = 64; st > 0; st >>= 1){
      if (tid < st && sv[tid+st] < sv[tid]){ sv[tid] = sv[tid+st]; si[tid] = si[tid+st]; }
      __syncthreads();
    }
    float mv = sv[0]; int mi = si[0];
    __syncthreads();
    if (tid == mi) v = 3.4e38f;                   // remove extracted element
    if (tid == 0){ if (t == 0) lmin = mv; lsum += mv; lsq += mv*mv; lmax = mv; }
  }
  if (tid == 0){
    atomicAdd(&accS[b], lsum);
    atomicAdd(&accQ[b], lsq);
    atomicMin(&accMn[b], __float_as_uint(lmin));  // dist>=0 -> uint order == float order
    atomicMax(&accMx[b], __float_as_uint(lmax));
  }
}

// ---------------- stats -> landscape MLP -> topo embedding (one block) ----------------
__global__ __launch_bounds__(256) void k_stats_topo(
    const float* accS, const float* accQ, const unsigned int* accMn, const unsigned int* accMx,
    const float* __restrict__ w1, const float* __restrict__ b1,
    const float* __restrict__ w2, const float* __restrict__ b2,
    const float* __restrict__ wd0, const float* __restrict__ bd0,
    const float* __restrict__ topoW, const float* __restrict__ topoB,
    const float* __restrict__ gate, float* __restrict__ topoOut){
  const int tid = threadIdx.x;
  __shared__ float st[BB][6];
  __shared__ float h1[BB][192];
  __shared__ float h2[BB][32];
  __shared__ float ld[BB][32];
  if (tid < BB){
    const float N = (float)(SS*KNN);
    float s = accS[tid], q = accQ[tid];
    float mean = s / N;
    float var  = (q - s*s/N) / (N - 1.0f);        // torch.std unbiased
    float sd   = sqrtf(fmaxf(var, 0.0f));
    st[tid][0] = mean; st[tid][1] = sd;
    st[tid][2] = __uint_as_float(accMn[tid]);
    st[tid][3] = __uint_as_float(accMx[tid]);
    st[tid][4] = mean*0.5f; st[tid][5] = sd*0.5f;
  }
  __syncthreads();
  for (int i = tid; i < BB*192; i += 256){
    int b = i / 192, j = i % 192;
    float a = b1[j];
    #pragma unroll
    for (int r = 0; r < 6; ++r) a += st[b][r] * w1[r*192 + j];
    h1[b][j] = fmaxf(a, 0.0f);
  }
  __syncthreads();
  for (int i = tid; i < BB*32; i += 256){
    int b = i / 32, j = i % 32;
    float a = b2[j];
    for (int r = 0; r < 192; ++r) a += h1[b][r] * w2[r*32 + j];
    h2[b][j] = a;
  }
  __syncthreads();
  for (int i = tid; i < BB*32; i += 256){
    int b = i / 32, j = i % 32;
    float a = bd0[j];
    #pragma unroll
    for (int r = 0; r < 32; ++r) a += h2[b][r] * wd0[r*32 + j];
    ld[b][j] = a;
  }
  __syncthreads();
  const float gt = gate[0];
  for (int i = tid; i < BB*DD; i += 256){
    int b = i / DD, d = i % DD;
    float a = topoB[d];
    #pragma unroll
    for (int r = 0; r < 32; ++r) a += ld[b][r] * topoW[r*DD + d];
    topoOut[(size_t)b*DD + d] = gt * a;           // pre-scale by gate
  }
}

// ---------------- attention: one wave per (b, h, 16-row q tile) ----------------
__global__ __launch_bounds__(32) void k_attn(const unsigned short* __restrict__ qh,
                                             const unsigned short* __restrict__ kh,
                                             const unsigned short* __restrict__ vT,
                                             unsigned short* __restrict__ o){
  const int tile = blockIdx.x;
  const int qt = tile % (SS/16);
  const int h  = (tile / (SS/16)) % HH;
  const int b  = tile / ((SS/16)*HH);
  const int lane = threadIdx.x;
  const int c = lane & 15, half = lane >> 4;

  __shared__ float          sc[16][SS];  // 32 KB scores/probs f32
  __shared__ unsigned short pb[16][SS];  // 16 KB probs bf16

  const unsigned short* Q  = qh + ((size_t)(b*HH + h)*SS)*HDD;
  const unsigned short* Kb = kh + ((size_t)(b*HH + h)*SS)*HDD;
  const unsigned short* Vp = vT + ((size_t)(b*HH + h)*HDD)*SS;

  // --- scores = (Q Kt) * 1/sqrt(64) ---
  const unsigned short* Arow = Q + (size_t)(qt*16 + c)*HDD;
  v16bf a0 = load_a_frag(Arow, 0, half);
  v16bf a1 = load_a_frag(Arow, 32, half);
  for (int nt = 0; nt < SS/16; ++nt){
    v8f acc = {};
    const unsigned short* B0 = Kb + (size_t)(nt*16 + c)*HDD + half*16;
    acc = wmma_bf16(a0, load_b_frag(B0),      acc);
    acc = wmma_bf16(a1, load_b_frag(B0 + 32), acc);
    #pragma unroll
    for (int r = 0; r < 8; ++r) sc[r + 8*half][nt*16 + c] = acc[r] * 0.125f;
  }
  __syncthreads();

  // --- softmax: 2 lanes per row ---
  {
    int r = lane >> 1, seg = lane & 1, c0 = seg * (SS/2);
    float mx = -3.4e38f;
    for (int cc = c0; cc < c0 + SS/2; ++cc) mx = fmaxf(mx, sc[r][cc]);
    mx = fmaxf(mx, __shfl_xor(mx, 1, 32));
    float sum = 0.f;
    for (int cc = c0; cc < c0 + SS/2; ++cc){ float e = __expf(sc[r][cc] - mx); sc[r][cc] = e; sum += e; }
    sum += __shfl_xor(sum, 1, 32);
    float inv = 1.0f / sum;
    for (int cc = c0; cc < c0 + SS/2; ++cc) pb[r][cc] = f2bf(sc[r][cc] * inv);
  }
  __syncthreads();

  // --- o = P @ V  (A from LDS bf16, B = vT rows) ---
  for (int nt = 0; nt < HDD/16; ++nt){
    v8f acc = {};
    for (int kt = 0; kt < SS; kt += 32){
      Frag16 a;
      a.q[0] = *(const uint4*)&pb[c][kt + 16*half];
      a.q[1] = *(const uint4*)&pb[c][kt + 16*half + 8];
      const unsigned short* B0 = Vp + (size_t)(nt*16 + c)*SS + kt + half*16;
      acc = wmma_bf16(a.v, load_b_frag(B0), acc);
    }
    #pragma unroll
    for (int r = 0; r < 8; ++r){
      int s = qt*16 + r + 8*half;
      o[((size_t)(b*SS + s))*DD + h*HDD + nt*16 + c] = f2bf(acc[r]);
    }
  }
}

// ---------------- LayerNorm (biased var + 1e-5, like jnp.var) ----------------
__global__ __launch_bounds__(256) void k_ln(const float* __restrict__ in,
                                            const float* __restrict__ g,
                                            const float* __restrict__ bta,
                                            float* __restrict__ outF,
                                            unsigned short* __restrict__ outB){
  int row = blockIdx.x, tid = threadIdx.x;
  const float* r = in + (size_t)row*DD;
  __shared__ float s1[256], s2[256];
  float a0 = 0.f, a1 = 0.f;
  for (int d = tid; d < DD; d += 256){ float v = r[d]; a0 += v; a1 += v*v; }
  s1[tid] = a0; s2[tid] = a1; __syncthreads();
  for (int st = 128; st > 0; st >>= 1){
    if (tid < st){ s1[tid] += s1[tid+st]; s2[tid] += s2[tid+st]; }
    __syncthreads();
  }
  float mean = s1[0] / (float)DD;
  float var  = s2[0] / (float)DD - mean*mean;
  float rstd = rsqrtf(var + 1e-5f);
  for (int d = tid; d < DD; d += 256){
    float y = (r[d] - mean) * rstd * g[d] + bta[d];
    outF[(size_t)row*DD + d] = y;
    if (outB) outB[(size_t)row*DD + d] = f2bf(y);
  }
}

// ---------------- host launcher ----------------
extern "C" void kernel_launch(void* const* d_in, const int* in_sizes, int n_in,
                              void* d_out, int out_size, void* d_ws, size_t ws_size,
                              hipStream_t stream){
  (void)in_sizes; (void)n_in; (void)out_size; (void)ws_size;
  const float* x     = (const float*)d_in[0];
  const int*   sidx  = (const int*)d_in[1];
  const float* w1    = (const float*)d_in[2];
  const float* b1    = (const float*)d_in[3];
  const float* w2    = (const float*)d_in[4];
  const float* b2    = (const float*)d_in[5];
  const float* wd0   = (const float*)d_in[6];
  const float* bd0   = (const float*)d_in[7];
  // d_in[8], d_in[9] (wd1, bd1) are unused by the reference
  const float* Wqkv  = (const float*)d_in[10];
  const float* bqkv  = (const float*)d_in[11];
  const float* Wout  = (const float*)d_in[12];
  const float* bout  = (const float*)d_in[13];
  const float* topoW = (const float*)d_in[14];
  const float* topoB = (const float*)d_in[15];
  const float* gate  = (const float*)d_in[16];
  const float* ln1g  = (const float*)d_in[17];
  const float* ln1b  = (const float*)d_in[18];
  const float* ln2g  = (const float*)d_in[19];
  const float* ln2b  = (const float*)d_in[20];
  const float* Wf1   = (const float*)d_in[21];
  const float* bf1   = (const float*)d_in[22];
  const float* Wf2   = (const float*)d_in[23];
  const float* bf2   = (const float*)d_in[24];
  float* out = (float*)d_out;

  char* ws = (char*)d_ws;
  size_t off = 0;
  auto take = [&](size_t bytes)->char*{
    char* p = ws + off; off = (off + bytes + 255) & ~(size_t)255; return p;
  };
  unsigned short* xb    = (unsigned short*)take((size_t)MM*DD*2);
  float*          xsq   = (float*)take((size_t)MM*4);
  unsigned short* sb    = (unsigned short*)take((size_t)BB*NSMP*DD*2);
  float*          ssq   = (float*)take((size_t)BB*NSMP*4);
  unsigned short* WqkvT = (unsigned short*)take((size_t)D3*DD*2);
  unsigned short* WoutT = (unsigned short*)take((size_t)DD*DD*2);
  unsigned short* Wf1T  = (unsigned short*)take((size_t)DFF*DD*2);
  unsigned short* Wf2T  = (unsigned short*)take((size_t)DD*DFF*2);
  float*          dist  = (float*)take((size_t)MM*NSMP*4);
  float*          accS  = (float*)take(64);
  float*          accQ  = (float*)take(64);
  unsigned int*   accMn = (unsigned int*)take(64);
  unsigned int*   accMx = (unsigned int*)take(64);
  float*          topo  = (float*)take((size_t)BB*DD*4);
  unsigned short* qh    = (unsigned short*)take((size_t)BB*HH*SS*HDD*2);
  unsigned short* kh    = (unsigned short*)take((size_t)BB*HH*SS*HDD*2);
  unsigned short* vT    = (unsigned short*)take((size_t)BB*HH*HDD*SS*2);
  unsigned short* ob    = (unsigned short*)take((size_t)MM*DD*2);
  float*          res1  = (float*)take((size_t)MM*DD*4);   // x + attn + topo
  float*          x1f   = (float*)take((size_t)MM*DD*4);
  unsigned short* x1b   = (unsigned short*)take((size_t)MM*DD*2);
  unsigned short* hffn  = (unsigned short*)take((size_t)MM*DFF*2);
  float*          res2  = res1;                             // res1 dead after LN1

  // ---- prep ----
  k_init_acc<<<1, 32, 0, stream>>>(accS, accQ, accMn, accMx);
  k_cvt_x<<<MM, 256, 0, stream>>>(x, xb, xsq);
  k_gather_s<<<BB*NSMP, 256, 0, stream>>>(xb, xsq, sidx, sb, ssq);
  {
    auto tgrid = [](size_t n){ return dim3((unsigned)((n + 255)/256)); };
    k_transpose<<<tgrid((size_t)DD*D3),  256, 0, stream>>>(Wqkv, WqkvT, DD, D3);
    k_transpose<<<tgrid((size_t)DD*DD),  256, 0, stream>>>(Wout, WoutT, DD, DD);
    k_transpose<<<tgrid((size_t)DD*DFF), 256, 0, stream>>>(Wf1,  Wf1T,  DD, DFF);
    k_transpose<<<tgrid((size_t)DFF*DD), 256, 0, stream>>>(Wf2,  Wf2T,  DFF, DD);
  }

  // ---- LSH distances: dist = sqrt(|x|^2 + |s|^2 - 2 x.s) ----
  {
    GemmP p = {};
    p.A = xb; p.Bt = sb; p.Kdim = DD; p.lda = DD; p.ldb = DD;
    p.xsq = xsq; p.ssq = ssq; p.dist = dist;
    k_gemm<4><<<dim3(MM/16, NSMP/64), 32, 0, stream>>>(p);
  }
  k_topk<<<MM, 128, 0, stream>>>(dist, accS, accQ, accMn, accMx);
  k_stats_topo<<<1, 256, 0, stream>>>(accS, accQ, accMn, accMx,
                                      w1, b1, w2, b2, wd0, bd0, topoW, topoB, gate, topo);

  // ---- QKV projection ----
  {
    GemmP p = {};
    p.A = xb; p.Bt = WqkvT; p.Kdim = DD; p.lda = DD; p.ldb = DD;
    p.bias = bqkv; p.q = qh; p.k = kh; p.v = vT;
    k_gemm<0><<<dim3(MM/16, D3/64), 32, 0, stream>>>(p);
  }

  // ---- attention ----
  k_attn<<<BB*HH*(SS/16), 32, 0, stream>>>(qh, kh, vT, ob);

  // ---- out-proj + bias + x + gate*topo -> res1 ----
  {
    GemmP p = {};
    p.A = ob; p.Bt = WoutT; p.Kdim = DD; p.lda = DD; p.ldb = DD;
    p.bias = bout; p.add = x; p.topo = topo; p.outF = res1;
    k_gemm<1><<<dim3(MM/16, DD/64), 32, 0, stream>>>(p);
  }
  k_ln<<<MM, 256, 0, stream>>>(res1, ln1g, ln1b, x1f, x1b);

  // ---- FFN ----
  {
    GemmP p = {};
    p.A = x1b; p.Bt = Wf1T; p.Kdim = DD; p.lda = DD; p.ldb = DD;
    p.bias = bf1; p.outB = hffn;
    k_gemm<2><<<dim3(MM/16, DFF/64), 32, 0, stream>>>(p);
  }
  {
    GemmP p = {};
    p.A = hffn; p.Bt = Wf2T; p.Kdim = DFF; p.lda = DFF; p.ldb = DFF;
    p.bias = bf2; p.add = x1f; p.outF = res2;
    k_gemm<3><<<dim3(MM/16, DD/64), 32, 0, stream>>>(p);
  }
  k_ln<<<MM, 256, 0, stream>>>(res2, ln2g, ln2b, out, (unsigned short*)nullptr);
}
